// DynamicGraphEncoder_85985245266279
// MI455X (gfx1250) — 8-bit matrix operand order repaired and checked by reading, compiled, not run on hardware
//
#include <hip/hip_runtime.h>
#include <hip/hip_bf16.h>
#include <math.h>
#include <stdint.h>

#define NND   1536
#define VDIM  128
#define NMOL  64
#define NEDGE 16384
#define TAU   0.2f
#define GAMMA 1.0f

typedef __attribute__((ext_vector_type(16))) int   v16i;
typedef __attribute__((ext_vector_type(8)))  float v8f;

#if __has_builtin(__builtin_amdgcn_global_load_async_to_lds_b128) && \
    __has_builtin(__builtin_amdgcn_s_wait_asynccnt)
#define USE_ASYNC_Q 1
typedef int v4i_vs __attribute__((vector_size(16)));
typedef v4i_vs __attribute__((address_space(1)))* gptr128_t;
typedef v4i_vs __attribute__((address_space(3)))* lptr128_t;
#else
#define USE_ASYNC_Q 0
#endif

__global__ void zero_scalars_kernel(float* __restrict__ out) {
    if (threadIdx.x < 3) out[2048 + threadIdx.x] = 0.0f;
}

__global__ __launch_bounds__(256) void cvt_fp8_kernel(
        const float* __restrict__ nem, uint32_t* __restrict__ nem8) {
    const size_t idx = (size_t)blockIdx.x * 256 + threadIdx.x;
    const float* src = nem + idx * 16;
    uint32_t w[4];
#pragma unroll
    for (int j = 0; j < 4; ++j) {
        uint32_t v = 0;
#pragma unroll
        for (int b = 0; b < 4; ++b)
            v |= (src[j * 4 + b] != 0.0f ? 0x38u : 0x00u) << (8 * b);
        w[j] = v;
    }
    uint4 o; o.x = w[0]; o.y = w[1]; o.z = w[2]; o.w = w[3];
    *(uint4*)(nem8 + idx * 4) = o;
}

__global__ __launch_bounds__(32) void pq_kernel(
        const float* __restrict__ v,
        const float* __restrict__ Wp, const float* __restrict__ bp,
        const float* __restrict__ Wq, const float* __restrict__ bq,
        float* __restrict__ p, float* __restrict__ q) {
    const int node = blockIdx.x, t = threadIdx.x;
    __shared__ float vrow[VDIM];
    for (int d = t; d < VDIM; d += 32) vrow[d] = v[node * VDIM + d];
    __syncthreads();
    const float* W = (t < 16) ? Wp : Wq;
    const float* b = (t < 16) ? bp : bq;
    const int col = t & 15;
    float acc = b[col];
    for (int d = 0; d < VDIM; ++d) acc += vrow[d] * W[d * 16 + col];
    if (t < 16) p[node * 16 + col] = acc;
    else        q[node * 16 + col] = acc;
}

__global__ __launch_bounds__(256) void molinfo_kernel(
        const float* __restrict__ mnm, int* __restrict__ mol_id,
        float* __restrict__ molsize) {
    const int i = blockIdx.x * 256 + threadIdx.x;
    if (i < NND) {
        int mid = 0;
        for (int k = 0; k < NMOL; ++k)
            if (mnm[k * NND + i] != 0.0f) mid = k;
        mol_id[i] = mid;
    }
    if (i < NMOL) {
        float s = 0.0f;
        for (int n = 0; n < NND; ++n) s += mnm[i * NND + n];
        molsize[i] = s;
    }
}

__global__ __launch_bounds__(64) void s_kernel(
        const float* __restrict__ mnm, const float* __restrict__ q,
        float* __restrict__ S) {
    const int mol = blockIdx.x, t = threadIdx.x;
    __shared__ float part[64][16];
    float acc[16];
#pragma unroll
    for (int d = 0; d < 16; ++d) acc[d] = 0.0f;
    for (int n = t; n < NND; n += 64) {
        const float wgt = mnm[mol * NND + n];
        if (wgt != 0.0f) {
#pragma unroll
            for (int d = 0; d < 16; ++d) acc[d] += wgt * q[n * 16 + d];
        }
    }
#pragma unroll
    for (int d = 0; d < 16; ++d) part[t][d] = acc[d];
    __syncthreads();
    if (t < 16) {
        float s = 0.0f;
        for (int r = 0; r < 64; ++r) s += part[r][t];
        S[mol * 16 + t] = s;
    }
}

__global__ __launch_bounds__(256) void cnorm_kernel(
        const float* __restrict__ S, float* __restrict__ closs) {
    __shared__ float red[256];
    const int t = threadIdx.x;
    float s = 0.0f;
    for (int i = t; i < NMOL * 16; i += 256) { float v = S[i]; s += v * v; }
    red[t] = s; __syncthreads();
    for (int off = 128; off > 0; off >>= 1) {
        if (t < off) red[t] += red[t + off];
        __syncthreads();
    }
    if (t == 0) closs[0] += sqrtf(red[0]);
}

__global__ __launch_bounds__(32) void layer_kernel(
        const float* __restrict__ S, const int* __restrict__ mol_id,
        const float* __restrict__ molsize,
        const float* __restrict__ Wc, const float* __restrict__ bc,
        const float* __restrict__ W1, const float* __restrict__ b1,
        const float* __restrict__ Wdp, const float* __restrict__ bdp,
        const float* __restrict__ Wdq, const float* __restrict__ bdq,
        float* __restrict__ p, float* __restrict__ q) {
    const int node = blockIdx.x, t = threadIdx.x;
    __shared__ float in[48];
    __shared__ float h[32];
    const int   mid = mol_id[node];
    const float cnt = molsize[mid];
    if (t < 16) {
        in[t]      = p[node * 16 + t];
        in[16 + t] = q[node * 16 + t];
    }
    __syncthreads();
    if (t < 16) {
        float acc = cnt * bc[t];
        for (int s = 0; s < 16; ++s)
            acc += (S[mid * 16 + s] - cnt * in[16 + s]) * Wc[s * 16 + t];
        in[32 + t] = acc;
    }
    __syncthreads();
    {
        float acc = b1[t];
        for (int d = 0; d < 48; ++d) acc += in[d] * W1[d * 32 + t];
        h[t] = acc > 0.0f ? acc : 0.0f;
    }
    __syncthreads();
    if (t < 16) {
        float acc = bdp[t];
        for (int s = 0; s < 32; ++s) acc += h[s] * Wdp[s * 16 + t];
        p[node * 16 + t] = in[t] + TAU * tanhf(acc);
    } else {
        const int d = t - 16;
        float acc = bdq[d];
        for (int s = 0; s < 32; ++s) acc += h[s] * Wdq[s * 16 + d];
        q[node * 16 + d] = in[16 + d] + TAU * tanhf(acc);
    }
}

__global__ __launch_bounds__(256) void w_kernel(
        const float* __restrict__ p, const float* __restrict__ q,
        const float* __restrict__ Wa, const float* __restrict__ ba,
        float* __restrict__ w, float* __restrict__ sloss) {
    const int i = blockIdx.x * 256 + threadIdx.x;
    __shared__ float red[256];
    float sabs = 0.0f;
    if (i < NND) {
        float acc = ba[0];
        for (int d = 0; d < 16; ++d) {
            const float pv = p[i * 16 + d];
            acc  += pv * Wa[d] + q[i * 16 + d] * Wa[16 + d];
            sabs += fabsf(pv);
        }
        w[i] = acc;
    }
    red[threadIdx.x] = sabs; __syncthreads();
    for (int off = 128; off > 0; off >>= 1) {
        if (threadIdx.x < off) red[threadIdx.x] += red[threadIdx.x + off];
        __syncthreads();
    }
    if (threadIdx.x == 0) atomicAdd(sloss, red[0]);
}

__global__ __launch_bounds__(128) void att_kernel(
        const float* __restrict__ mnm, const float* __restrict__ mask,
        const float* __restrict__ w,
        const float* __restrict__ p, const float* __restrict__ q,
        float* __restrict__ f) {
    const int mol = blockIdx.x, t = threadIdx.x;
    __shared__ float red[128];
    __shared__ float smax, sdenom;
    float mx = -3.0e38f;
    for (int n = t; n < NND; n += 128) {
        const float lg = mnm[mol * NND + n] * w[n] + mask[mol * NND + n];
        mx = fmaxf(mx, lg);
    }
    red[t] = mx; __syncthreads();
    for (int off = 64; off > 0; off >>= 1) {
        if (t < off) red[t] = fmaxf(red[t], red[t + off]);
        __syncthreads();
    }
    if (t == 0) smax = red[0];
    __syncthreads();
    mx = smax;
    float se = 0.0f, xa[32];
#pragma unroll
    for (int d = 0; d < 32; ++d) xa[d] = 0.0f;
    for (int n = t; n < NND; n += 128) {
        const float lg = mnm[mol * NND + n] * w[n] + mask[mol * NND + n];
        const float ex = expf(lg - mx);
        se += ex;
        if (ex > 0.0f) {
#pragma unroll
            for (int d = 0; d < 16; ++d) {
                xa[d]      += ex * p[n * 16 + d];
                xa[16 + d] += ex * q[n * 16 + d];
            }
        }
    }
    __syncthreads();
    red[t] = se; __syncthreads();
    for (int off = 64; off > 0; off >>= 1) {
        if (t < off) red[t] += red[t + off];
        __syncthreads();
    }
    if (t == 0) sdenom = red[0];
    __syncthreads();
    const float inv = 1.0f / sdenom;
    for (int d = 0; d < 32; ++d) {
        __syncthreads();
        red[t] = xa[d]; __syncthreads();
        for (int off = 64; off > 0; off >>= 1) {
            if (t < off) red[t] += red[t + off];
            __syncthreads();
        }
        if (t == 0) f[mol * 32 + d] = red[0] * inv;
    }
}

__global__ __launch_bounds__(128) void aloss_kernel(
        const uint8_t* __restrict__ nem8, const float* __restrict__ q,
        float* __restrict__ aloss) {
    const int K  = NEDGE;
    const int NB = NND / 32;
    const int wave = blockIdx.x * 4 + (threadIdx.x >> 5);
    const int lane = threadIdx.x & 31;

    int bi = 0, trem = wave;
    while (trem >= NB - bi) { trem -= NB - bi; ++bi; }
    const int bj = bi + trem;
    const int I0 = bi * 32, J0 = bj * 32;

    const int half = lane >> 4;
    const int r    = lane & 15;
    const int n    = lane & 15;

#if USE_ASYNC_Q
    __shared__ float qlds[4][64][16];
    const int wv = threadIdx.x >> 5;
    {
#pragma unroll
        for (int j = 0; j < 8; ++j) {
            const int c = lane + 32 * j;
            const int row = c >> 2, qt = c & 3;
            const int grow = (row < 32) ? (I0 + row) : (J0 + row - 32);
            const float* gp = q + (size_t)grow * 16 + qt * 4;
            float* lp = &qlds[wv][row][qt * 4];
            __builtin_amdgcn_global_load_async_to_lds_b128(
                (gptr128_t)gp, (lptr128_t)lp, 0, 0);
        }
    }
#endif

    v8f acc[2][2] = {};
    const uint8_t* rowA0 = nem8 + (size_t)(I0 + r)      * K;
    const uint8_t* rowA1 = nem8 + (size_t)(I0 + 16 + r) * K;
    const uint8_t* rowB0 = nem8 + (size_t)(J0 + r)      * K;
    const uint8_t* rowB1 = nem8 + (size_t)(J0 + 16 + r) * K;

    for (int k = 0; k < K; k += 128) {
        v16i a0, a1, b0, b1;
#pragma unroll
        for (int u = 0; u < 4; ++u) {
            const int off = k + 16 * u + 8 * half;
            int2 lo0 = *(const int2*)(rowA0 + off);
            int2 hi0 = *(const int2*)(rowA0 + off + 64);
            int2 lo1 = *(const int2*)(rowA1 + off);
            int2 hi1 = *(const int2*)(rowA1 + off + 64);
            a0[2*u] = lo0.x; a0[2*u+1] = lo0.y; a0[8+2*u] = hi0.x; a0[9+2*u] = hi0.y;
            a1[2*u] = lo1.x; a1[2*u+1] = lo1.y; a1[8+2*u] = hi1.x; a1[9+2*u] = hi1.y;
        }
#pragma unroll
        for (int u = 0; u < 4; ++u) {
            const int off = k + 16 * u + 8 * half;
            int2 lo0 = *(const int2*)(rowB0 + off);
            int2 hi0 = *(const int2*)(rowB0 + off + 64);
            int2 lo1 = *(const int2*)(rowB1 + off);
            int2 hi1 = *(const int2*)(rowB1 + off + 64);
            b0[2*u] = lo0.x; b0[2*u+1] = lo0.y; b0[8+2*u] = hi0.x; b0[9+2*u] = hi0.y;
            b1[2*u] = lo1.x; b1[2*u+1] = lo1.y; b1[8+2*u] = hi1.x; b1[9+2*u] = hi1.y;
        }
        acc[0][0] = __builtin_amdgcn_wmma_f32_16x16x128_fp8_fp8(a0, b0, (short)0,
                        acc[0][0], false, false);
        acc[0][1] = __builtin_amdgcn_wmma_f32_16x16x128_fp8_fp8(a0, b1, (short)0,
                        acc[0][1], false, false);
        acc[1][0] = __builtin_amdgcn_wmma_f32_16x16x128_fp8_fp8(a1, b0, (short)0,
                        acc[1][0], false, false);
        acc[1][1] = __builtin_amdgcn_wmma_f32_16x16x128_fp8_fp8(a1, b1, (short)0,
                        acc[1][1], false, false);
        if (k + 128 < K) {
            __builtin_prefetch(rowA0 + k + 128 + 8 * half, 0, 1);
            __builtin_prefetch(rowB0 + k + 128 + 16 * half, 0, 1);
        }
    }

#if USE_ASYNC_Q
    __builtin_amdgcn_s_wait_asynccnt(0);
    __syncthreads();
#define QI_PTR(m)    (&qlds[wv][(m)][0])
#define QJ_PTR(b,nn) (&qlds[wv][32 + 16 * (b) + (nn)][0])
#else
#define QI_PTR(m)    (q + (size_t)(I0 + (m)) * 16)
#define QJ_PTR(b,nn) (q + (size_t)(J0 + 16 * (b) + (nn)) * 16)
#endif

    float qj0r[16], qj1r[16];
#pragma unroll
    for (int d = 0; d < 16; ++d) {
        qj0r[d] = QJ_PTR(0, n)[d];
        qj1r[d] = QJ_PTR(1, n)[d];
    }
    float lsum = 0.0f;
#pragma unroll
    for (int a = 0; a < 2; ++a) {
#pragma unroll
        for (int vr = 0; vr < 8; ++vr) {
            const int m  = 16 * a + 8 * half + vr;
            const int gi = I0 + m;
            float qir[16];
#pragma unroll
            for (int d = 0; d < 16; ++d) qir[d] = QI_PTR(m)[d];
#pragma unroll
            for (int b = 0; b < 2; ++b) {
                const int gj = J0 + 16 * b + n;
                if (gj > gi) {
                    const float* qjr = b ? qj1r : qj0r;
                    float sq = 0.0f;
#pragma unroll
                    for (int d = 0; d < 16; ++d) {
                        const float df = qir[d] - qjr[d];
                        sq += df * df;
                    }
                    const float nrm = sq > 0.0f ? sqrtf(sq) : 0.0f;
                    const float pen = nrm - GAMMA;
                    if (pen > 0.0f) lsum += 2.0f * acc[a][b][vr] * pen;
                }
            }
        }
    }
    for (int off = 16; off > 0; off >>= 1) lsum += __shfl_down(lsum, off, 32);
    if (lane == 0) atomicAdd(aloss, lsum);
}

extern "C" void kernel_launch(void* const* d_in, const int* in_sizes, int n_in,
                              void* d_out, int out_size, void* d_ws, size_t ws_size,
                              hipStream_t stream) {
    const float* v_features = (const float*)d_in[0];
    const float* mnm        = (const float*)d_in[1];
    const float* mmask      = (const float*)d_in[2];
    const float* nem        = (const float*)d_in[3];
    const float* Wp  = (const float*)d_in[5];
    const float* bp  = (const float*)d_in[6];
    const float* Wq  = (const float*)d_in[7];
    const float* bq  = (const float*)d_in[8];
    const float* Wc  = (const float*)d_in[9];
    const float* bc  = (const float*)d_in[10];
    const float* W1  = (const float*)d_in[11];
    const float* b1  = (const float*)d_in[12];
    const float* Wdp = (const float*)d_in[13];
    const float* bdp = (const float*)d_in[14];
    const float* Wdq = (const float*)d_in[15];
    const float* bdq = (const float*)d_in[16];
    const float* Wa  = (const float*)d_in[17];
    const float* ba  = (const float*)d_in[18];

    float* out   = (float*)d_out;
    float* sloss = out + 2048;
    float* closs = out + 2049;
    float* aloss = out + 2050;

    uint8_t* nem8 = (uint8_t*)d_ws;
    float* fws    = (float*)d_ws + (size_t)NND * NEDGE / 4;
    float* p       = fws;
    float* q       = fws + 24576;
    float* S       = fws + 49152;
    float* w       = fws + 50176;
    float* molsize = fws + 51712;
    int*   mol_id  = (int*)(fws + 51776);

    zero_scalars_kernel<<<1, 32, 0, stream>>>(out);
    cvt_fp8_kernel<<<(NND * NEDGE) / (16 * 256), 256, 0, stream>>>(nem, (uint32_t*)nem8);
    pq_kernel<<<NND, 32, 0, stream>>>(v_features, Wp, bp, Wq, bq, p, q);
    molinfo_kernel<<<6, 256, 0, stream>>>(mnm, mol_id, molsize);

    for (int layer = 0; layer < 3; ++layer) {
        s_kernel<<<NMOL, 64, 0, stream>>>(mnm, q, S);
        cnorm_kernel<<<1, 256, 0, stream>>>(S, closs);
        layer_kernel<<<NND, 32, 0, stream>>>(S, mol_id, molsize, Wc, bc,
                                             W1, b1, Wdp, bdp, Wdq, bdq, p, q);
    }
    s_kernel<<<NMOL, 64, 0, stream>>>(mnm, q, S);
    cnorm_kernel<<<1, 256, 0, stream>>>(S, closs);

    w_kernel<<<6, 256, 0, stream>>>(p, q, Wa, ba, w, sloss);
    att_kernel<<<NMOL, 128, 0, stream>>>(mnm, mmask, w, p, q, out);
    aloss_kernel<<<294, 128, 0, stream>>>(nem8, q, aloss);
}
